// DeepseekMoE_45183055954090
// MI455X (gfx1250) — compile-verified, matrix operand order checked by reading, not run on hardware
//
#include <hip/hip_runtime.h>
#include <hip/hip_bf16.h>
#include <stdint.h>

// ---------------- problem constants (DeepSeek-V2-Lite-like MoE) -------------
#define T_TOK 1024
#define HDIM  2048
#define NEXP  64
#define KTOP  6
#define IDIM  1024
#define NGRP  8
#define TOPG  3
#define CAP   256
#define SDIM  2048      // NS * I
#define RSFAC 2.5f

typedef __bf16 bf16x16 __attribute__((ext_vector_type(16)));
typedef float  f32x8   __attribute__((ext_vector_type(8)));
typedef unsigned int u32x4 __attribute__((ext_vector_type(4)));
typedef int          i32x8 __attribute__((ext_vector_type(8)));
typedef int          i32x4 __attribute__((ext_vector_type(4)));

__device__ __forceinline__ unsigned short f2bf(float x) {
  unsigned u = __float_as_uint(x);
  unsigned r = u + 0x7fffu + ((u >> 16) & 1u);   // round-to-nearest-even
  return (unsigned short)(r >> 16);
}
__device__ __forceinline__ float bf2f(unsigned short h) {
  return __uint_as_float(((unsigned)h) << 16);
}

// ---- Tensor Data Mover: 2-D tile load global->LDS (ISA ch.7/8, D# packing) --
// data_size_code: 0=1B 1=2B 2=4B 3=8B. tile_d0 = elements per row,
// tile_d1 = rows, stride0 = row stride (in elements). LDS receives the tile
// densely row-major (tile_d0*elt bytes per row, rows contiguous).
__device__ __forceinline__ void tdm_load_2d(unsigned lds_addr, const void* gptr,
                                            unsigned data_size_code,
                                            unsigned tile_d0, unsigned tile_d1,
                                            unsigned long long stride0) {
  unsigned long long ga = (unsigned long long)gptr;
  u32x4 g0;
  g0[0] = 1u;                                      // count=1, user descriptor
  g0[1] = lds_addr;                                // D# lds_addr
  g0[2] = (unsigned)(ga & 0xffffffffu);            // global_addr[31:0]
  g0[3] = (unsigned)((ga >> 32) & 0x01ffffffu)     // global_addr[56:32]
        | (2u << 30);                              // type = 2 ("image")
  unsigned long long td0 = tile_d0, td1 = tile_d1; // tensor dims = tile dims
  i32x8 g1;
  g1[0] = (int)(data_size_code << 16);             // wg_mask=0, flags=0
  g1[1] = (int)((unsigned)(td0 & 0xffffu) << 16);  // tensor_dim0[15:0]
  g1[2] = (int)(((unsigned)(td0 >> 16) & 0xffffu) |
                ((unsigned)(td1 & 0xffffu) << 16));          // td0 hi | td1 lo
  g1[3] = (int)(((unsigned)(td1 >> 16) & 0xffffu) |
                ((tile_d0 & 0xffffu) << 16));                // td1 hi | tile_dim0
  g1[4] = (int)(tile_d1 & 0xffffu);                // tile_dim1 | tile_dim2=0
  g1[5] = (int)(stride0 & 0xffffffffull);          // tensor_dim0_stride lo
  g1[6] = (int)((stride0 >> 32) & 0xffffull);      // stride hi | dim1_stride=0
  g1[7] = 0;
  i32x4 z4 = {0, 0, 0, 0};
#if __clang_major__ >= 23
  i32x8 z8 = {0, 0, 0, 0, 0, 0, 0, 0};
  __builtin_amdgcn_tensor_load_to_lds(g0, g1, z4, z4, z8, 0);
#else
  __builtin_amdgcn_tensor_load_to_lds(g0, g1, z4, z4, 0);
#endif
}
__device__ __forceinline__ unsigned lds_off(const void* p) {
  return (unsigned)(uintptr_t)p;   // LDS aperture: flat addr[31:0] == LDS offset
}

// ======================= 1) router: sigmoid + grouped top-k =================
__global__ __launch_bounds__(64)
void router_kernel(const float* __restrict__ x, const float* __restrict__ gate_w,
                   const float* __restrict__ gate_bias,
                   float* __restrict__ topk_w, int* __restrict__ topk_ids) {
  __shared__ __align__(16) float xs[HDIM];
  __shared__ float sig[NEXP];
  __shared__ float scb[NEXP];
  const int t = blockIdx.x, tid = threadIdx.x;

  const float4* xr = (const float4*)(x + (size_t)t * HDIM);
  for (int i = tid; i < HDIM / 4; i += 64) ((float4*)xs)[i] = xr[i];
  __syncthreads();

  const float* w = gate_w + (size_t)tid * HDIM;
  float s = 0.f;
  for (int h = 0; h < HDIM; h += 4) {
    float4 wv = *(const float4*)(w + h);
    s += xs[h] * wv.x + xs[h + 1] * wv.y + xs[h + 2] * wv.z + xs[h + 3] * wv.w;
  }
  float sg = 1.f / (1.f + __expf(-s));
  sig[tid] = sg;
  scb[tid] = sg + gate_bias[tid];
  __syncthreads();

  if (tid == 0) {
    float g2[NGRP];
#pragma unroll
    for (int g = 0; g < NGRP; ++g) {
      float m1 = -1e30f, m2 = -1e30f;
#pragma unroll
      for (int j = 0; j < NEXP / NGRP; ++j) {
        float v = scb[g * (NEXP / NGRP) + j];
        if (v > m1) { m2 = m1; m1 = v; } else if (v > m2) m2 = v;
      }
      g2[g] = m1 + m2;
    }
    unsigned gmask = 0;
#pragma unroll
    for (int it = 0; it < TOPG; ++it) {
      int best = 0; float bv = -1e30f;
#pragma unroll
      for (int g = 0; g < NGRP; ++g)
        if (!((gmask >> g) & 1u) && g2[g] > bv) { bv = g2[g]; best = g; }
      gmask |= 1u << best;
    }
    unsigned long long used = 0ull;
    int ids[KTOP]; float wv[KTOP]; float wsum = 0.f;
#pragma unroll
    for (int kk = 0; kk < KTOP; ++kk) {
      int best = 0; float bv = -1e30f;
      for (int e = 0; e < NEXP; ++e) {
        if ((used >> e) & 1ull) continue;
        if (!((gmask >> (e / (NEXP / NGRP))) & 1u)) continue;
        float v = scb[e];
        if (v > bv) { bv = v; best = e; }
      }
      used |= 1ull << best;
      ids[kk] = best; wv[kk] = sig[best]; wsum += sig[best];
    }
    float inv = RSFAC / (wsum + 1e-20f);
#pragma unroll
    for (int kk = 0; kk < KTOP; ++kk) {
      topk_ids[t * KTOP + kk] = ids[kk];
      topk_w[t * KTOP + kk] = wv[kk] * inv;
    }
  }
}

// ============ 2) deterministic dispatch (reference cumsum order) ============
__global__ __launch_bounds__(64)
void dispatch_kernel(const int* __restrict__ topk_ids, const float* __restrict__ topk_w,
                     int* __restrict__ tok, float* __restrict__ slotw) {
  const int e = threadIdx.x;
  int cnt = 0;
  for (int i = 0; i < T_TOK * KTOP; ++i) {
    if (topk_ids[i] == e) {
      if (cnt < CAP) { tok[e * CAP + cnt] = i / KTOP; slotw[e * CAP + cnt] = topk_w[i]; }
      ++cnt;
    }
  }
  int c = cnt < CAP ? cnt : CAP;
  for (int p = c; p < CAP; ++p) { tok[e * CAP + p] = -1; slotw[e * CAP + p] = 0.f; }
}

// ========= 3) gate_up GEMM + SiluAndMul -> bf16 act (routed & shared) =======
// block: 256 rows x (64 gate cols + 64 up cols); waves 0-3 gate, 4-7 up.
// Weights pulled by TDM into LDS raw (f32), transposed+converted in LDS.
__global__ __launch_bounds__(256)
void gateup_kernel(const float* __restrict__ X, const int* __restrict__ tok,
                   const float* __restrict__ W, int Ipar,
                   unsigned short* __restrict__ act) {
  __shared__ __align__(16) unsigned short Ap[256 * 32];  // [row][k] bf16
  __shared__ __align__(16) unsigned short Wp[128 * 32];  // [col][k] bf16
  __shared__ __align__(16) float          Wraw[2 * 32 * 64]; // [reg][k][c] f32 (TDM)
  __shared__ __align__(16) unsigned short Up[256 * 64];  // up-acc exchange

  const int tid = threadIdx.x, lane = tid & 31, wvid = tid >> 5;
  const int rowbase = blockIdx.x * 256;
  const int j0 = blockIdx.y * 64;
  const size_t ldW = 2 * (size_t)Ipar;
  const float* We = W + (tok ? (size_t)blockIdx.x * (size_t)HDIM * ldW : 0);

  f32x8 acc[16];
#pragma unroll
  for (int i = 0; i < 16; ++i)
#pragma unroll
    for (int e = 0; e < 8; ++e) acc[i][e] = 0.f;

  const int ar = lane & 15;
  const int ko = (lane < 16) ? 0 : 8;
  const int cb = wvid * 16 + ar;

  for (int k0 = 0; k0 < HDIM; k0 += 32) {
    __syncthreads();   // previous iteration fully consumed LDS
    if (wvid == 0) {   // TDM: gate panel + up panel (32 rows x 64 f32, stride ldW)
      tdm_load_2d(lds_off(&Wraw[0]),       We + (size_t)k0 * ldW + j0,
                  /*f32*/2, 64, 32, (unsigned long long)ldW);
      tdm_load_2d(lds_off(&Wraw[32 * 64]), We + (size_t)k0 * ldW + Ipar + j0,
                  /*f32*/2, 64, 32, (unsigned long long)ldW);
    }
    {  // stage A: gathered x rows -> bf16 (zeros for empty slots), overlaps TDM
      int r = tid >> 3, c4 = (tid & 7) * 4;
      for (int rr = r; rr < 256; rr += 32) {
        int trow = tok ? tok[rowbase + rr] : (rowbase + rr);
        unsigned short h0 = 0, h1 = 0, h2 = 0, h3 = 0;
        if (trow >= 0) {
          const float* src = X + (size_t)trow * HDIM + k0 + c4;
          float4 v = *(const float4*)src;
          if (k0 + 32 < HDIM) __builtin_prefetch(src + 32, 0, 1);
          h0 = f2bf(v.x); h1 = f2bf(v.y); h2 = f2bf(v.z); h3 = f2bf(v.w);
        }
        *(uint2*)&Ap[rr * 32 + c4] =
            make_uint2((unsigned)h0 | ((unsigned)h1 << 16),
                       (unsigned)h2 | ((unsigned)h3 << 16));
      }
    }
    if (wvid == 0) __builtin_amdgcn_s_wait_tensorcnt(0);
    __syncthreads();   // TDM data + A panel visible to all waves
#pragma unroll
    for (int i = 0; i < 4; ++i) {  // LDS f32 -> LDS bf16 transposed [col][k]
      int q = tid + i * 256;                 // quad id, 1024 quads
      int reg = q >> 9, kk = (q >> 4) & 31, c4 = (q & 15) * 4;
      const float4 v = *(const float4*)&Wraw[reg * 2048 + kk * 64 + c4];
      int cbase = reg * 64 + c4;
      Wp[(cbase + 0) * 32 + kk] = f2bf(v.x);
      Wp[(cbase + 1) * 32 + kk] = f2bf(v.y);
      Wp[(cbase + 2) * 32 + kk] = f2bf(v.z);
      Wp[(cbase + 3) * 32 + kk] = f2bf(v.w);
    }
    __syncthreads();

    union { uint4 q[2]; bf16x16 v; } bf;
    bf.q[0] = *(const uint4*)&Wp[cb * 32 + ko];
    bf.q[1] = *(const uint4*)&Wp[cb * 32 + ko + 16];
#pragma unroll
    for (int mt = 0; mt < 16; ++mt) {
      union { uint4 q[2]; bf16x16 v; } af;
      const unsigned short* ap = &Ap[(mt * 16 + ar) * 32 + ko];
      af.q[0] = *(const uint4*)ap;
      af.q[1] = *(const uint4*)(ap + 16);
      acc[mt] = __builtin_amdgcn_wmma_f32_16x16x32_bf16(
          false, af.v, false, bf.v, (short)0, acc[mt], false, false);
    }
  }

  __syncthreads();
  const int mbase = (lane < 16) ? 0 : 8;
  if (wvid >= 4) {  // publish up-halves
    int c = (wvid - 4) * 16 + ar;
#pragma unroll
    for (int mt = 0; mt < 16; ++mt)
#pragma unroll
      for (int v = 0; v < 8; ++v)
        Up[(mt * 16 + mbase + v) * 64 + c] = f2bf(acc[mt][v]);
  }
  __syncthreads();
  if (wvid < 4) {  // silu(gate) * up -> act (bf16)
    int c = wvid * 16 + ar;
#pragma unroll
    for (int mt = 0; mt < 16; ++mt)
#pragma unroll
      for (int v = 0; v < 8; ++v) {
        int row = mt * 16 + mbase + v;
        float g = acc[mt][v];
        float u = bf2f(Up[row * 64 + c]);
        float a = (g / (1.f + __expf(-g))) * u;
        act[(size_t)(rowbase + row) * Ipar + (j0 + c)] = f2bf(a);
      }
  }
}

// ============== 4) down GEMM + combine (shared store / routed atomic) =======
// A panel (bf16 act) is a pure TDM copy; W2 panel TDM'd raw then converted.
__global__ __launch_bounds__(256)
void down_kernel(const unsigned short* __restrict__ act, const float* __restrict__ W2,
                 int Kd, const int* __restrict__ tok, const float* __restrict__ slotw,
                 float* __restrict__ out) {
  __shared__ __align__(16) unsigned short Ap[256 * 32];   // [row][k] bf16 (TDM)
  __shared__ __align__(16) unsigned short Wp[128 * 32];   // [col][k] bf16
  __shared__ __align__(16) float          Wraw[32 * 128]; // [k][c] f32 (TDM)

  const int tid = threadIdx.x, lane = tid & 31, wvid = tid >> 5;
  const int rowbase = blockIdx.x * 256;
  const int j0 = blockIdx.y * 128;
  const float* We = W2 + (tok ? (size_t)blockIdx.x * (size_t)Kd * HDIM : 0);

  f32x8 acc[16];
#pragma unroll
  for (int i = 0; i < 16; ++i)
#pragma unroll
    for (int e = 0; e < 8; ++e) acc[i][e] = 0.f;

  const int ar = lane & 15;
  const int ko = (lane < 16) ? 0 : 8;
  const int cb = wvid * 16 + ar;

  for (int k0 = 0; k0 < Kd; k0 += 32) {
    __syncthreads();
    if (wvid == 0) {
      // act tile: 256 rows x 32 bf16, row stride Kd elements
      tdm_load_2d(lds_off(&Ap[0]), act + (size_t)rowbase * Kd + k0,
                  /*2B*/1, 32, 256, (unsigned long long)Kd);
      // W2 tile: 32 rows x 128 f32, row stride HDIM elements
      tdm_load_2d(lds_off(&Wraw[0]), We + (size_t)k0 * HDIM + j0,
                  /*4B*/2, 128, 32, (unsigned long long)HDIM);
      __builtin_amdgcn_s_wait_tensorcnt(0);
    }
    __syncthreads();
#pragma unroll
    for (int i = 0; i < 4; ++i) {  // LDS f32 -> LDS bf16 transposed [col][k]
      int q = tid + i * 256;                 // 1024 quads
      int kk = q >> 5, c4 = (q & 31) * 4;
      const float4 v = *(const float4*)&Wraw[kk * 128 + c4];
      Wp[(c4 + 0) * 32 + kk] = f2bf(v.x);
      Wp[(c4 + 1) * 32 + kk] = f2bf(v.y);
      Wp[(c4 + 2) * 32 + kk] = f2bf(v.z);
      Wp[(c4 + 3) * 32 + kk] = f2bf(v.w);
    }
    __syncthreads();

    union { uint4 q[2]; bf16x16 v; } bf;
    bf.q[0] = *(const uint4*)&Wp[cb * 32 + ko];
    bf.q[1] = *(const uint4*)&Wp[cb * 32 + ko + 16];
#pragma unroll
    for (int mt = 0; mt < 16; ++mt) {
      union { uint4 q[2]; bf16x16 v; } af;
      const unsigned short* ap = &Ap[(mt * 16 + ar) * 32 + ko];
      af.q[0] = *(const uint4*)ap;
      af.q[1] = *(const uint4*)(ap + 16);
      acc[mt] = __builtin_amdgcn_wmma_f32_16x16x32_bf16(
          false, af.v, false, bf.v, (short)0, acc[mt], false, false);
    }
  }

  const int c = j0 + wvid * 16 + ar;
  const int mbase = (lane < 16) ? 0 : 8;
  if (tok) {  // routed: weighted atomic combine
#pragma unroll
    for (int mt = 0; mt < 16; ++mt)
#pragma unroll
      for (int v = 0; v < 8; ++v) {
        int slot = rowbase + mt * 16 + mbase + v;
        int t = tok[slot];
        if (t >= 0)
          __hip_atomic_fetch_add(out + (size_t)t * HDIM + c, slotw[slot] * acc[mt][v],
                                 __ATOMIC_RELAXED, __HIP_MEMORY_SCOPE_AGENT);
      }
  } else {  // shared expert: plain store (initializes full out)
#pragma unroll
    for (int mt = 0; mt < 16; ++mt)
#pragma unroll
      for (int v = 0; v < 8; ++v)
        out[(size_t)(rowbase + mt * 16 + mbase + v) * HDIM + c] = acc[mt][v];
  }
}

// ================================ launcher ==================================
extern "C" void kernel_launch(void* const* d_in, const int* in_sizes, int n_in,
                              void* d_out, int out_size, void* d_ws, size_t ws_size,
                              hipStream_t stream) {
  const float* x   = (const float*)d_in[0];
  const float* gw  = (const float*)d_in[1];
  const float* gb  = (const float*)d_in[2];
  const float* w13 = (const float*)d_in[3];
  const float* w2  = (const float*)d_in[4];
  const float* sgu = (const float*)d_in[5];
  const float* sdn = (const float*)d_in[6];
  float* out = (float*)d_out;

  // workspace carve-up (~38 MB total)
  char* ws = (char*)d_ws;
  size_t off = 0;
  auto carve = [&](size_t bytes) -> void* {
    void* p = ws + off;
    off = (off + bytes + 255) & ~(size_t)255;
    return p;
  };
  float* topkw = (float*)carve(T_TOK * KTOP * sizeof(float));
  int*   topki = (int*)  carve(T_TOK * KTOP * sizeof(int));
  int*   tok   = (int*)  carve(NEXP * CAP * sizeof(int));
  float* slw   = (float*)carve(NEXP * CAP * sizeof(float));
  unsigned short* actR = (unsigned short*)carve((size_t)NEXP * CAP * IDIM * 2);
  unsigned short* actS = (unsigned short*)carve((size_t)T_TOK * SDIM * 2);

  router_kernel<<<T_TOK, 64, 0, stream>>>(x, gw, gb, topkw, topki);
  dispatch_kernel<<<1, 64, 0, stream>>>(topki, topkw, tok, slw);

  // routed experts: gate_up + silu*up (one block per expert per 64-col strip)
  gateup_kernel<<<dim3(NEXP, IDIM / 64), 256, 0, stream>>>(x, tok, w13, IDIM, actR);
  // shared expert
  gateup_kernel<<<dim3(T_TOK / 256, SDIM / 64), 256, 0, stream>>>(x, nullptr, sgu, SDIM, actS);

  // shared down first: stores => fully initializes out
  down_kernel<<<dim3(T_TOK / 256, HDIM / 128), 256, 0, stream>>>(actS, sdn, SDIM,
                                                                 nullptr, nullptr, out);
  // routed down: weighted atomic accumulate
  down_kernel<<<dim3(NEXP, HDIM / 128), 256, 0, stream>>>(actR, w2, IDIM, tok, slw, out);
}
